// ROIHead_81750407512653
// MI455X (gfx1250) — matrix operand order repaired and checked by reading, not compiled, not run on hardware
//
#include <hip/hip_runtime.h>
#include <hip/hip_bf16.h>

typedef __attribute__((ext_vector_type(16))) _Float16 v16h;
typedef __attribute__((ext_vector_type(8)))  _Float16 v8h;
typedef __attribute__((ext_vector_type(4)))  _Float16 v4h;
typedef __attribute__((ext_vector_type(8)))  float    v8f;

#define N_ROIS      2000
#define NUM_CLASSES 81
#define IN_CH       25088
#define N_GT        64
#define HID         1024
#define N_LOC       (NUM_CLASSES * 4)   // 324
#define IOU_BG_THRESH 0.4f
#define NMS_THRESH    0.5f

// ---------------------------------------------------------------------------
// Tiled WMMA GEMM: C[M,N] = act(A[M,K] @ B[K,N] + bias)
// A: f32 (AHALF=0) or f16 (AHALF=1). C: f32 (OHALF=0) or f16 (OHALF=1).
// Block tile 128x128, K-step 64, 256 threads = 8 waves, wave tile 32x64.
// Double-buffered LDS software pipeline + global prefetch one staged tile ahead.
// B staged via in-register 4x4 transpose -> vector ds_store_b64 (no b16 scatter).
// ---------------------------------------------------------------------------
template<int AHALF, int OHALF>
__global__ __launch_bounds__(256)
void gemm_wmma(const void* __restrict__ Av, const float* __restrict__ B,
               const float* __restrict__ bias, void* __restrict__ Cv,
               int M, int N, int K, int relu)
{
    constexpr int BM = 128, BN = 128, BK = 64;
    constexpr int LDT = 72;                    // padded k-stride (halfs): 144B, 16B-aligned
    __shared__ _Float16 sA[2][BM * LDT];       // row-major  [m][k]
    __shared__ _Float16 sB[2][BN * LDT];       // transposed [n][k]

    const float*    Af = (const float*)Av;
    const _Float16* Ah = (const _Float16*)Av;

    const int tid  = threadIdx.x;
    const int lane = tid & 31;
    const int wave = tid >> 5;
    const int bm   = blockIdx.x * BM;
    const int bn   = blockIdx.y * BN;
    const int wrow = (wave & 3) * 32;          // wave 32 rows
    const int wcol = (wave >> 2) * 64;         // wave 64 cols

    const v8f zacc = {};
    v8f acc[2][4];
#pragma unroll
    for (int r = 0; r < 2; ++r)
#pragma unroll
        for (int c = 0; c < 4; ++c) acc[r][c] = zacc;

    // ---- staging thread mapping ----
    // A: 128 rows x 64 k; thread covers rows arow+32*i (i=0..3), 8 consecutive k
    const int arow = tid >> 3;
    const int acol = (tid & 7) * 8;
    // B: 64 k x 128 n; thread covers k quads bk4+i (+32 second pass), 4 consecutive n
    const int bcol = (tid & 31) * 4;
    const int bk4  = (tid >> 5) * 4;

    const float*    aPf[4];
    const _Float16* aPh[4];
    bool            aOk[4];
#pragma unroll
    for (int i = 0; i < 4; ++i) {
        int gr  = bm + arow + 32 * i;
        aOk[i]  = (gr < M);
        int grc = aOk[i] ? gr : (M - 1);
        aPf[i]  = Af + (size_t)grc * K + acol;
        aPh[i]  = Ah + (size_t)grc * K + acol;
    }
    const float* bP[4];
#pragma unroll
    for (int i = 0; i < 4; ++i)
        bP[i] = B + (size_t)(bk4 + i) * N + bn + bcol;

    const int  gcB   = bn + bcol;
    const bool bFull = (gcB + 3 < N);
    const int  nk    = K / BK;

    v8h aReg[4];                // A rows, 8 halfs each
    v4h bReg[2][4];             // B 4x4 blocks (converted), [pass][k-in-quad]
    const v8h zh8 = {};

    auto loadTile = [&]() {
#pragma unroll
        for (int i = 0; i < 4; ++i) {
            v8h h;
            if constexpr (AHALF) {
                h = *(const v8h*)aPh[i];
            } else {
                float4 f0 = *(const float4*)aPf[i];
                float4 f1 = *(const float4*)(aPf[i] + 4);
                h = (v8h){(_Float16)f0.x, (_Float16)f0.y, (_Float16)f0.z, (_Float16)f0.w,
                          (_Float16)f1.x, (_Float16)f1.y, (_Float16)f1.z, (_Float16)f1.w};
            }
            if (!aOk[i]) h = zh8;
            aReg[i] = h;
            aPf[i] += BK;
            aPh[i] += BK;
        }
#pragma unroll
        for (int i2 = 0; i2 < 2; ++i2)
#pragma unroll
            for (int i = 0; i < 4; ++i) {
                const float* src = bP[i] + (size_t)(32 * i2) * N;
                v4h h;
                if (bFull) {
                    float4 f4 = *(const float4*)src;
                    h = (v4h){(_Float16)f4.x, (_Float16)f4.y, (_Float16)f4.z, (_Float16)f4.w};
                } else {
                    float t[4];
#pragma unroll
                    for (int j = 0; j < 4; ++j)
                        t[j] = (gcB + j < N) ? src[j] : 0.f;
                    h = (v4h){(_Float16)t[0], (_Float16)t[1], (_Float16)t[2], (_Float16)t[3]};
                }
                bReg[i2][i] = h;
            }
#pragma unroll
        for (int i = 0; i < 4; ++i)
            bP[i] += (size_t)BK * N;
    };

    auto storeLDS = [&](int buf) {
#pragma unroll
        for (int i = 0; i < 4; ++i)
            *(v8h*)&sA[buf][(arow + 32 * i) * LDT + acol] = aReg[i];   // ds_store_b128
#pragma unroll
        for (int i2 = 0; i2 < 2; ++i2) {
            const int kb = bk4 + 32 * i2;
#pragma unroll
            for (int j = 0; j < 4; ++j) {
                // register 4x4 transpose: 4 consecutive k for column bcol+j
                v4h h = (v4h){bReg[i2][0][j], bReg[i2][1][j],
                              bReg[i2][2][j], bReg[i2][3][j]};
                *(v4h*)&sB[buf][(bcol + j) * LDT + kb] = h;            // ds_store_b64
            }
        }
    };

    // fragment addressing per CDNA5 16-bit WMMA layouts
    const int mf = wrow + (lane & 15);
    const int kA = (lane >> 4) * 8;            // 0 or 8
    const int nf = wcol + (lane & 15);
    const int kB = (lane >> 4) * 8;            // 0 or 8

    auto compute = [&](int buf) {
#pragma unroll
        for (int ks = 0; ks < 2; ++ks) {       // two 32-wide K sub-steps per staged tile
            const int kloc = 32 * ks;
            union { v16h v; v8h h[2]; } af[2], bf[4];
#pragma unroll
            for (int r = 0; r < 2; ++r) {
                const _Float16* base = &sA[buf][(mf + 16 * r) * LDT + kloc + kA];
                af[r].h[0] = *(const v8h*)base;
                af[r].h[1] = *(const v8h*)(base + 16);
            }
#pragma unroll
            for (int c = 0; c < 4; ++c) {
                const _Float16* base = &sB[buf][(nf + 16 * c) * LDT + kloc + kB];
                bf[c].h[0] = *(const v8h*)base;
                bf[c].h[1] = *(const v8h*)(base + 16);
            }
#pragma unroll
            for (int r = 0; r < 2; ++r)
#pragma unroll
                for (int c = 0; c < 4; ++c)
                    acc[r][c] = __builtin_amdgcn_wmma_f32_16x16x32_f16(
                        false, af[r].v, false, bf[c].v, (short)0, acc[r][c], false, false);
        }
    };

    // ---- software pipeline ----
    loadTile();
    storeLDS(0);
    __syncthreads();
    for (int kt = 0; kt < nk; ++kt) {
        const int cur = kt & 1;
        if (kt + 1 < nk) {
            loadTile();                         // tile kt+1 -> registers
            if (kt + 2 < nk) {                  // prefetch tile kt+2 (ptrs already advanced)
                if constexpr (AHALF) __builtin_prefetch((const void*)aPh[0], 0, 1);
                else                 __builtin_prefetch((const void*)aPf[0], 0, 1);
                __builtin_prefetch((const void*)bP[0], 0, 1);
            }
        }
        compute(cur);                           // wmma on tile kt
        if (kt + 1 < nk) storeLDS(cur ^ 1);
        __syncthreads();
    }

    // ---- C store: lane n = lane&15, VGPR r -> row r + 8*(lane>>4) ----
    const int crl = (lane >> 4) * 8;
    const int ccl = lane & 15;
#pragma unroll
    for (int r = 0; r < 2; ++r) {
        int growb = bm + wrow + 16 * r + crl;
#pragma unroll
        for (int c = 0; c < 4; ++c) {
            int gc = bn + wcol + 16 * c + ccl;
            if (gc >= N) continue;
            float bv = bias[gc];
#pragma unroll
            for (int rr = 0; rr < 8; ++rr) {
                int gr = growb + rr;
                if (gr >= M) continue;
                float v = acc[r][c][rr] + bv;
                if (relu) v = fmaxf(v, 0.f);
                if (OHALF) ((_Float16*)Cv)[(size_t)gr * N + gc] = (_Float16)v;
                else       ((float*)Cv)[(size_t)gr * N + gc] = v;
            }
        }
    }
}

// ---------------------------------------------------------------------------
// rois (x1,y1,x2,y2) -> (y1,x1,h,w)
// ---------------------------------------------------------------------------
__global__ void prep_rois(const float* __restrict__ rois, float* __restrict__ yxhw)
{
    int i = blockIdx.x * blockDim.x + threadIdx.x;
    if (i >= N_ROIS) return;
    float x1 = rois[i*4+0], y1 = rois[i*4+1], x2 = rois[i*4+2], y2 = rois[i*4+3];
    yxhw[i*4+0] = y1; yxhw[i*4+1] = x1; yxhw[i*4+2] = y2 - y1; yxhw[i*4+3] = x2 - x1;
}

// loc = loc_all[i, gt_anchor_label[i], :]
__global__ void select_loc(const float* __restrict__ loc_all,
                           const int* __restrict__ labels,
                           float* __restrict__ loc_sel)
{
    int i = blockIdx.x * blockDim.x + threadIdx.x;
    if (i >= N_ROIS) return;
    int c = labels[i];
    for (int j = 0; j < 4; ++j)
        loc_sel[i*4+j] = loc_all[i*N_LOC + c*4 + j];
}

// single-block mean / inv-std (ddof=1) over 8000 values
__global__ __launch_bounds__(1024)
void stats_kernel(const float* __restrict__ loc, float* __restrict__ stats)
{
    __shared__ float ss[1024], sq[1024];
    int t = threadIdx.x;
    float s = 0.f, q = 0.f;
    for (int i = t; i < N_ROIS * 4; i += 1024) {
        float v = loc[i]; s += v; q += v * v;
    }
    ss[t] = s; sq[t] = q; __syncthreads();
    for (int o = 512; o > 0; o >>= 1) {
        if (t < o) { ss[t] += ss[t+o]; sq[t] += sq[t+o]; }
        __syncthreads();
    }
    if (t == 0) {
        float n = (float)(N_ROIS * 4);
        float mean = ss[0] / n;
        float var  = (sq[0] - ss[0] * ss[0] / n) / (n - 1.f);
        stats[0] = mean;
        stats[1] = rsqrtf(fmaxf(var, 1e-30f));
    }
}

// per-roi: standardize loc, softmax, loc2box, iou/argmax vs gt, box2loc, nms inputs
__global__ void finalize_kernel(const float* __restrict__ loc_sel,
                                const float* __restrict__ stats,
                                const float* __restrict__ cls_logit,
                                const float* __restrict__ rois_yxhw,
                                const float* __restrict__ gt_bbox,
                                const int*   __restrict__ gt_label,
                                float* __restrict__ loc_std,
                                float* __restrict__ cls_prob,
                                float* __restrict__ pred_box,
                                float* __restrict__ score,
                                float* __restrict__ iou_max_o,
                                float* __restrict__ gt_roi_loc,
                                int*   __restrict__ gt_roi_label,
                                float* __restrict__ nms_boxes)
{
    int i = blockIdx.x * blockDim.x + threadIdx.x;
    if (i >= N_ROIS) return;
    float mean = stats[0], istd = stats[1];
    float l0 = (loc_sel[i*4+0] - mean) * istd;
    float l1 = (loc_sel[i*4+1] - mean) * istd;
    float l2 = (loc_sel[i*4+2] - mean) * istd;
    float l3 = (loc_sel[i*4+3] - mean) * istd;
    loc_std[i*4+0] = l0; loc_std[i*4+1] = l1; loc_std[i*4+2] = l2; loc_std[i*4+3] = l3;

    // softmax over 81 logits (two-pass, register-light)
    const float* lg = cls_logit + (long long)i * NUM_CLASSES;
    float mx = lg[0];
    for (int c = 1; c < NUM_CLASSES; ++c) mx = fmaxf(mx, lg[c]);
    float sum = 0.f;
    for (int c = 0; c < NUM_CLASSES; ++c) sum += __expf(lg[c] - mx);
    float inv = 1.f / sum;
    float best = 0.f;
    for (int c = 0; c < NUM_CLASSES; ++c) {
        float p = __expf(lg[c] - mx) * inv;
        cls_prob[(long long)i * NUM_CLASSES + c] = p;
        best = fmaxf(best, p);
    }
    score[i] = best;

    // loc2box
    float sy = rois_yxhw[i*4+0], sx = rois_yxhw[i*4+1];
    float sh = rois_yxhw[i*4+2], sw = rois_yxhw[i*4+3];
    float cy = sy + 0.5f * sh, cx = sx + 0.5f * sw;
    float pcy = l0 * sh + cy, pcx = l1 * sw + cx;
    float ph = sh * __expf(l2), pw = sw * __expf(l3);
    float pby = pcy - 0.5f * ph, pbx = pcx - 0.5f * pw;
    pred_box[i*4+0] = pby; pred_box[i*4+1] = pbx;
    pred_box[i*4+2] = ph;  pred_box[i*4+3] = pw;

    // IoU vs 64 gt boxes (y,x,h,w), argmax (first max)
    float ay1 = pby, ax1 = pbx, ay2 = pby + ph, ax2 = pbx + pw;
    float areaA = ph * pw;
    float best_iou = -3.0e38f; int bidx = 0;
    for (int g = 0; g < N_GT; ++g) {
        float by1 = gt_bbox[g*4+0], bx1 = gt_bbox[g*4+1];
        float bh  = gt_bbox[g*4+2], bw  = gt_bbox[g*4+3];
        float ih = fmaxf(fminf(ay2, by1 + bh) - fmaxf(ay1, by1), 0.f);
        float iw = fmaxf(fminf(ax2, bx1 + bw) - fmaxf(ax1, bx1), 0.f);
        float inter = ih * iw;
        float iou = inter / (areaA + bh * bw - inter);
        if (iou > best_iou) { best_iou = iou; bidx = g; }
    }
    iou_max_o[i] = best_iou;

    // box2loc against matched gt
    float d0 = gt_bbox[bidx*4+0], d1 = gt_bbox[bidx*4+1];
    float d2 = gt_bbox[bidx*4+2], d3 = gt_bbox[bidx*4+3];
    float gcy = d0 + 0.5f * d2, gcx = d1 + 0.5f * d3;
    gt_roi_loc[i*4+0] = (gcy - cy) / sh;
    gt_roi_loc[i*4+1] = (gcx - cx) / sw;
    gt_roi_loc[i*4+2] = __logf(d2 / sh);
    gt_roi_loc[i*4+3] = __logf(d3 / sw);
    gt_roi_label[i] = (best_iou < IOU_BG_THRESH) ? 0 : gt_label[bidx];

    // corners for NMS: [x, y, x+w, y+h]
    nms_boxes[i*4+0] = pbx;      nms_boxes[i*4+1] = pby;
    nms_boxes[i*4+2] = pbx + pw; nms_boxes[i*4+3] = pby + ph;
}

// ---------------------------------------------------------------------------
// single-block NMS: bitonic sort (desc by score) over 2048, greedy suppression
// ---------------------------------------------------------------------------
__global__ __launch_bounds__(1024)
void nms_kernel(const float* __restrict__ boxes, const float* __restrict__ scores,
                int* __restrict__ keep_out)
{
    constexpr int NB = 2048;
    __shared__ float s_sc[NB];
    __shared__ int   s_id[NB];
    __shared__ float s_x1[NB], s_y1[NB], s_x2[NB], s_y2[NB], s_ar[NB];
    __shared__ unsigned char s_keep[NB];
    const int t = threadIdx.x;

    for (int i = t; i < NB; i += 1024) {
        s_sc[i] = (i < N_ROIS) ? scores[i] : -3.0e38f;
        s_id[i] = i;
    }
    __syncthreads();

    // bitonic sort descending
    for (int k = 2; k <= NB; k <<= 1) {
        for (int j = k >> 1; j > 0; j >>= 1) {
            for (int i = t; i < NB; i += 1024) {
                int ixj = i ^ j;
                if (ixj > i) {
                    bool up = ((i & k) == 0);
                    float a = s_sc[i], b = s_sc[ixj];
                    bool sw = up ? (a < b) : (a > b);
                    if (sw) {
                        s_sc[i] = b; s_sc[ixj] = a;
                        int tmp = s_id[i]; s_id[i] = s_id[ixj]; s_id[ixj] = tmp;
                    }
                }
            }
            __syncthreads();
        }
    }

    // gather sorted boxes
    for (int i = t; i < NB; i += 1024) {
        int id = s_id[i];
        if (id < N_ROIS) {
            float x1 = boxes[id*4+0], y1 = boxes[id*4+1];
            float x2 = boxes[id*4+2], y2 = boxes[id*4+3];
            s_x1[i] = x1; s_y1[i] = y1; s_x2[i] = x2; s_y2[i] = y2;
            s_ar[i] = (x2 - x1) * (y2 - y1);
            s_keep[i] = 1;
        } else {
            s_x1[i] = 0.f; s_y1[i] = 0.f; s_x2[i] = 0.f; s_y2[i] = 0.f;
            s_ar[i] = 0.f; s_keep[i] = 0;
        }
    }
    __syncthreads();

    // greedy suppression (sequential over ranked boxes, parallel over victims)
    for (int i = 0; i < N_ROIS; ++i) {
        if (s_keep[i]) {
            float x1 = s_x1[i], y1 = s_y1[i], x2 = s_x2[i], y2 = s_y2[i], ar = s_ar[i];
            for (int j = i + 1 + t; j < NB; j += 1024) {
                if (s_keep[j]) {
                    float iw = fmaxf(fminf(x2, s_x2[j]) - fmaxf(x1, s_x1[j]), 0.f);
                    float ih = fmaxf(fminf(y2, s_y2[j]) - fmaxf(y1, s_y1[j]), 0.f);
                    float inter = iw * ih;
                    float iou = inter / (ar + s_ar[j] - inter);
                    if (iou > NMS_THRESH) s_keep[j] = 0;
                }
            }
        }
        __syncthreads();
    }

    for (int i = t; i < NB; i += 1024) {
        int id = s_id[i];
        if (id < N_ROIS) keep_out[id] = (int)s_keep[i];
    }
}

// ---------------------------------------------------------------------------
// masked final write: loc | cls | gt_roi_loc | gt_roi_label | pred_box | score
// ---------------------------------------------------------------------------
__global__ void write_out(const float* __restrict__ loc_std,
                          const float* __restrict__ cls_prob,
                          const float* __restrict__ gt_roi_loc,
                          const int*   __restrict__ gt_roi_label,
                          const float* __restrict__ pred_box,
                          const float* __restrict__ score,
                          const int*   __restrict__ keep,
                          float* __restrict__ out)
{
    int i = blockIdx.x * blockDim.x + threadIdx.x;
    if (i >= N_ROIS) return;
    const int O_LOC = 0, O_CLS = 8000, O_GRL = 170000, O_LBL = 178000,
              O_PB = 180000, O_SC = 188000;
    float m = keep[i] ? 1.f : 0.f;
    for (int j = 0; j < 4; ++j) out[O_LOC + i*4 + j] = loc_std[i*4+j] * m;
    for (int c = 0; c < NUM_CLASSES; ++c)
        out[O_CLS + (long long)i*NUM_CLASSES + c] = cls_prob[(long long)i*NUM_CLASSES + c] * m;
    for (int j = 0; j < 4; ++j) out[O_GRL + i*4 + j] = gt_roi_loc[i*4+j] * m;
    out[O_LBL + i] = (float)gt_roi_label[i] * m;
    for (int j = 0; j < 4; ++j) out[O_PB + i*4 + j] = pred_box[i*4+j] * m;
    out[O_SC + i] = score[i] * m;
}

// ---------------------------------------------------------------------------
extern "C" void kernel_launch(void* const* d_in, const int* in_sizes, int n_in,
                              void* d_out, int out_size, void* d_ws, size_t ws_size,
                              hipStream_t stream)
{
    const float* rpn_out         = (const float*)d_in[0];
    const float* rois            = (const float*)d_in[1];
    const int*   gt_anchor_label = (const int*)  d_in[2];
    const float* gt_bbox         = (const float*)d_in[3];
    const int*   gt_label        = (const int*)  d_in[4];
    const float* W1    = (const float*)d_in[5];
    const float* b1    = (const float*)d_in[6];
    const float* W2    = (const float*)d_in[7];
    const float* b2    = (const float*)d_in[8];
    const float* W_loc = (const float*)d_in[9];
    const float* b_loc = (const float*)d_in[10];
    const float* W_cls = (const float*)d_in[11];
    const float* b_cls = (const float*)d_in[12];

    // workspace carve (256-B aligned)
    char* p = (char*)d_ws;
    auto carve = [&](size_t bytes) -> void* {
        void* r = (void*)p;
        p += (bytes + 255) & ~(size_t)255;
        return r;
    };
    void*  h1         = carve((size_t)N_ROIS * HID * 2);          // f16
    void*  h2         = carve((size_t)N_ROIS * HID * 2);          // f16
    float* loc_all    = (float*)carve((size_t)N_ROIS * N_LOC * 4);
    float* cls_logit  = (float*)carve((size_t)N_ROIS * NUM_CLASSES * 4);
    float* cls_prob   = (float*)carve((size_t)N_ROIS * NUM_CLASSES * 4);
    float* rois_yxhw  = (float*)carve((size_t)N_ROIS * 4 * 4);
    float* loc_sel    = (float*)carve((size_t)N_ROIS * 4 * 4);
    float* loc_std    = (float*)carve((size_t)N_ROIS * 4 * 4);
    float* stats      = (float*)carve(256);
    float* pred_box   = (float*)carve((size_t)N_ROIS * 4 * 4);
    float* score      = (float*)carve((size_t)N_ROIS * 4);
    float* iou_max    = (float*)carve((size_t)N_ROIS * 4);
    float* gt_roi_loc = (float*)carve((size_t)N_ROIS * 4 * 4);
    int*   gt_roi_lab = (int*)  carve((size_t)N_ROIS * 4);
    float* nms_boxes  = (float*)carve((size_t)N_ROIS * 4 * 4);
    int*   keep       = (int*)  carve((size_t)N_ROIS * 4);

    const int TB = 256;
    const int roi_blocks = (N_ROIS + TB - 1) / TB;

    prep_rois<<<roi_blocks, TB, 0, stream>>>(rois, rois_yxhw);

    // GEMM1: h1 = relu(rpn_out @ W1 + b1), f32 in, f16 out
    gemm_wmma<0,1><<<dim3((N_ROIS+127)/128, (HID+127)/128), TB, 0, stream>>>(
        (const void*)rpn_out, W1, b1, h1, N_ROIS, HID, IN_CH, 1);
    // GEMM2: h2 = relu(h1 @ W2 + b2), f16 in, f16 out
    gemm_wmma<1,1><<<dim3((N_ROIS+127)/128, (HID+127)/128), TB, 0, stream>>>(
        h1, W2, b2, h2, N_ROIS, HID, HID, 1);
    // GEMM3: loc_all = h2 @ W_loc + b_loc  (N=324)
    gemm_wmma<1,0><<<dim3((N_ROIS+127)/128, (N_LOC+127)/128), TB, 0, stream>>>(
        h2, W_loc, b_loc, loc_all, N_ROIS, N_LOC, HID, 0);
    // GEMM4: cls_logit = h2 @ W_cls + b_cls  (N=81)
    gemm_wmma<1,0><<<dim3((N_ROIS+127)/128, 1), TB, 0, stream>>>(
        h2, W_cls, b_cls, cls_logit, N_ROIS, NUM_CLASSES, HID, 0);

    select_loc<<<roi_blocks, TB, 0, stream>>>(loc_all, gt_anchor_label, loc_sel);
    stats_kernel<<<1, 1024, 0, stream>>>(loc_sel, stats);
    finalize_kernel<<<roi_blocks, TB, 0, stream>>>(
        loc_sel, stats, cls_logit, rois_yxhw, gt_bbox, gt_label,
        loc_std, cls_prob, pred_box, score, iou_max, gt_roi_loc,
        gt_roi_lab, nms_boxes);
    nms_kernel<<<1, 1024, 0, stream>>>(nms_boxes, iou_max, keep);
    write_out<<<roi_blocks, TB, 0, stream>>>(
        loc_std, cls_prob, gt_roi_loc, gt_roi_lab, pred_box, score, keep,
        (float*)d_out);
}
